// CenterContrastiveLoss_40458591928682
// MI455X (gfx1250) — matrix operand order repaired and checked by reading, not compiled, not run on hardware
//
#include <hip/hip_runtime.h>
#include <stdint.h>

// ---------------- problem constants (match reference) ----------------
#define B_ 2048
#define C_ 65536
#define D_ 256
#define NNEG 50
#define EPS_ 0.1f
#define KSMOOTH 500.0f

// ---------------- tiling ----------------
#define RT 32                  // rows per workgroup tile (M tile): 2 WMMA row-sets
#define NRT (B_ / RT)          // 64 row tiles
#define CHUNK 1024             // columns per workgroup chunk (N tile)
#define NCHUNK (C_ / CHUNK)    // 64 chunks
#define PITCH (CHUNK + 20)     // LDS score row pitch (floats); kills bank conflicts
#define CANDCAP 128            // boundary-bin candidate cap per row
#define ROWS_PER_WAVE (RT / 8) // 4 rows per wave in selection phase
#define NEG_INF (-__builtin_inff())

// ---------------- WMMA / TDM types ----------------
typedef __attribute__((ext_vector_type(16))) __bf16 v16bf;
typedef __attribute__((ext_vector_type(8)))  float  v8f;
typedef __attribute__((ext_vector_type(4)))  unsigned int v4u_;
typedef __attribute__((ext_vector_type(8)))  int    v8i_;
typedef __attribute__((ext_vector_type(4)))  int    v4i_;

union BF16x16 { uint4 q[2]; v16bf v; };

#if __has_builtin(__builtin_amdgcn_tensor_load_to_lds)
#define HAVE_TDM 1
#endif

// ---------------- fp32 -> bf16 (round-to-nearest-even) ----------------
static __device__ __forceinline__ unsigned short f2bf(float x) {
    unsigned int u = __float_as_uint(x);
    u += 0x7FFFu + ((u >> 16) & 1u);
    return (unsigned short)(u >> 16);
}

// order-preserving uint mapping of float (monotone: a<b <=> ou(a)<ou(b))
static __device__ __forceinline__ unsigned ordu(float x) {
    unsigned u = __float_as_uint(x);
    return u ^ ((unsigned)((int)u >> 31) | 0x80000000u);
}

// =====================================================================
// K1: convert f [B,D] and centers [C,D] fp32 -> bf16 into workspace
// =====================================================================
__global__ void __launch_bounds__(256) k_convert(const float* __restrict__ f,
                                                 const float* __restrict__ centers,
                                                 unsigned short* __restrict__ fb,
                                                 unsigned short* __restrict__ cb) {
    const size_t nf = (size_t)B_ * D_ / 4;
    const size_t nc = (size_t)C_ * D_ / 4;
    const size_t stride = (size_t)gridDim.x * 256u;
    for (size_t q = (size_t)blockIdx.x * 256u + threadIdx.x; q < nf + nc; q += stride) {
        float4 v = (q < nf) ? ((const float4*)f)[q] : ((const float4*)centers)[q - nf];
        unsigned long long p = (unsigned long long)f2bf(v.x)
                             | ((unsigned long long)f2bf(v.y) << 16)
                             | ((unsigned long long)f2bf(v.z) << 32)
                             | ((unsigned long long)f2bf(v.w) << 48);
        if (q < nf) ((unsigned long long*)fb)[q] = p;
        else        ((unsigned long long*)cb)[q - nf] = p;
    }
}

// =====================================================================
// K2: fused bf16 WMMA GEMM (32 x CHUNK x 256) + histogram top-50
//   grid = (NRT, NCHUNK), block = 256 (8 waves)
//   Each B fragment feeds TWO WMMAs (rows 0-15 and 16-31) => half the
//   centers/L2 traffic per FLOP vs RT=16.
//   dynamic LDS (sA at offset 0: TDM descriptor uses lds_addr = 0):
//     sA (16384) | sLab (128) | sWCnt (128) | sCCnt (128) |
//     sHist (32*256*4) | sCand (32*CANDCAP*4) | sS (RT*PITCH*4)
// =====================================================================
#define OFF_LAB   (16384)
#define OFF_WCNT  (OFF_LAB + 128)
#define OFF_CCNT  (OFF_WCNT + 128)
#define OFF_HIST  (OFF_CCNT + 128)
#define OFF_CAND  (OFF_HIST + RT * 256 * 4)
#define OFF_S     (OFF_CAND + RT * CANDCAP * 4)
#define SMEM_K2   (OFF_S + RT * PITCH * 4)

__global__ void __launch_bounds__(256) k_gemm_topk(const unsigned short* __restrict__ fb,
                                                   const unsigned short* __restrict__ cb,
                                                   const int* __restrict__ label,
                                                   float* __restrict__ partial) {
    extern __shared__ char smem[];
    unsigned short* sA    = (unsigned short*)smem;          // 32 x 256 bf16 @ LDS 0
    int*            sLab  = (int*)(smem + OFF_LAB);
    int*            sWCnt = (int*)(smem + OFF_WCNT);
    int*            sCCnt = (int*)(smem + OFF_CCNT);
    int*            sHist = (int*)(smem + OFF_HIST);
    float*          sCand = (float*)(smem + OFF_CAND);
    float*          sS    = (float*)(smem + OFF_S);

    const int rowTile    = blockIdx.x;
    const int chunk      = blockIdx.y;
    const int tid        = threadIdx.x;
    const int wave       = tid >> 5;
    const int lane       = tid & 31;
    const int chunkStart = chunk * CHUNK;

    // ---- stage A tile (16 KB contiguous): TDM DMA if available ----
#ifdef HAVE_TDM
    if (tid == 0) {
        unsigned long long ga = (unsigned long long)(const void*)(fb + (size_t)rowTile * RT * D_);
        // D# group0: count=1 | lds_addr=0 | global_addr | type=2
        v4u_ g0 = { 1u, 0u, (unsigned)ga, (unsigned)(ga >> 32) | 0x80000000u };
        // D# group1: data_size=8B; tensor_dim0=2048; tensor_dim1=1; tile_dim0=2048;
        //            tile_dim1=1; tensor_dim0_stride=2048
        v8i_ g1 = { (int)(3u << 16),
                    (int)(2048u << 16),   // tensor_dim0 lo16 (bits 63:48)
                    (int)(1u << 16),      // tensor_dim1 lo16 (bits 95:80)
                    (int)(2048u << 16),   // tile_dim0 (bits 127:112)
                    1,                    // tile_dim1 (bits 143:128)
                    2048,                 // tensor_dim0_stride lo32
                    0, 0 };
        v4i_ z4 = {0, 0, 0, 0};
#if defined(__clang_major__) && __clang_major__ >= 23
        v8i_ z8 = {0, 0, 0, 0, 0, 0, 0, 0};
        __builtin_amdgcn_tensor_load_to_lds(g0, g1, z4, z4, z8, 0);
#else
        __builtin_amdgcn_tensor_load_to_lds(g0, g1, z4, z4, 0);
#endif
#if __has_builtin(__builtin_amdgcn_s_wait_tensorcnt)
        __builtin_amdgcn_s_wait_tensorcnt(0);
#else
        asm volatile("s_wait_tensorcnt 0" ::: "memory");
#endif
    }
#else
    {
        const uint4* src = (const uint4*)(fb + (size_t)rowTile * RT * D_);
        uint4*       dst = (uint4*)sA;
        for (int i = tid; i < (RT * D_ * 2) / 16; i += 256) dst[i] = src[i];
    }
#endif
    // ---- labels + zero histograms/counters ----
    if (tid < RT) { sLab[tid] = label[rowTile * RT + tid]; sWCnt[tid] = 0; sCCnt[tid] = 0; }
    for (int i = tid; i < RT * 256; i += 256) sHist[i] = 0;
    __syncthreads();

    // ---- hoist A fragments for BOTH 16-row sets into VGPRs (128 VGPRs) ----
    const int lh = lane & 15;
    const int kh = lane >> 4;
    v16bf af[2][8];
#pragma unroll
    for (int s = 0; s < 2; ++s) {
#pragma unroll
        for (int ks = 0; ks < 8; ++ks) {
            BF16x16 t;
            const char* base = (const char*)sA + (s * 16 + lh) * 512 + ks * 64 + kh * 16;
            t.q[0] = *(const uint4*)(base);
            t.q[1] = *(const uint4*)(base + 32);
            af[s][ks] = t.v;
        }
    }

    // ---- GEMM: each wave owns 8 of 64 tiles; each B frag -> 2 WMMAs ----
    for (int t = wave; t < CHUNK / 16; t += 8) {
        const int    n0   = t * 16;
        const size_t brow = (size_t)(chunkStart + n0 + lh) * D_;
        v8f c0 = {0.f, 0.f, 0.f, 0.f, 0.f, 0.f, 0.f, 0.f};
        v8f c1 = {0.f, 0.f, 0.f, 0.f, 0.f, 0.f, 0.f, 0.f};
#pragma unroll
        for (int ks = 0; ks < 8; ++ks) {
            BF16x16 bt;
            const unsigned short* bp = cb + brow + ks * 32 + kh * 8;
            bt.q[0] = *(const uint4*)(bp);
            bt.q[1] = *(const uint4*)(bp + 16);
            c0 = __builtin_amdgcn_wmma_f32_16x16x32_bf16(
                    false, af[0][ks], false, bt.v, (short)0, c0, false, false);
            c1 = __builtin_amdgcn_wmma_f32_16x16x32_bf16(
                    false, af[1][ks], false, bt.v, (short)0, c1, false, false);
        }
        if (t + 8 < CHUNK / 16)
            __builtin_prefetch(cb + (size_t)(chunkStart + (t + 8) * 16 + lh) * D_, 0, 1);

        const int colL = n0 + lh;
        const int colG = chunkStart + colL;
#pragma unroll
        for (int s = 0; s < 2; ++s) {
#pragma unroll
            for (int r = 0; r < 8; ++r) {
                const int rowL = s * 16 + r + 8 * kh;
                float v = (s == 0) ? c0[r] : c1[r];
                if (colG == sLab[rowL]) v = NEG_INF;      // mask positive column
                sS[rowL * PITCH + colL] = v;
                const int bin = (int)(ordu(v) >> 24);     // 256-bin sign/exp histogram
                atomicAdd(&sHist[rowL * 256 + bin], 1);
            }
        }
    }
    __syncthreads();

    // ---- histogram select: wave w handles 4 rows ----
    for (int rr = 0; rr < ROWS_PER_WAVE; ++rr) {
        const int rowL = wave * ROWS_PER_WAVE + rr;
        const int rowG = rowTile * RT + rowL;
        int*   H       = sHist + rowL * 256;
        float* rowS    = sS + rowL * PITCH;
        float* cand    = sCand + rowL * CANDCAP;
        float* outP    = partial + ((size_t)rowG * NCHUNK + chunk) * NNEG;

        // suffix counts over 256 bins: boundary bin b with
        //   W = #values in bins > b < 50 <= W + H[b]
        int seg = 0;
#pragma unroll
        for (int k = 0; k < 8; ++k) seg += H[lane * 8 + k];
        int run = seg;
#pragma unroll
        for (int off = 1; off < 32; off <<= 1) {
            int tt = __shfl_down(run, off, 32);
            if (lane + off < 32) run += tt;
        }
        int bloc = -1, Wloc = 0;
        if (run >= NNEG && (run - seg) < NNEG) {          // crossing lane (unique)
            int cum = run - seg;
            for (int bin = lane * 8 + 7; bin >= lane * 8; --bin) {
                const int h = H[bin];
                if (cum + h >= NNEG) { bloc = bin; Wloc = cum; break; }
                cum += h;
            }
        }
        unsigned long long msk = __ballot(bloc >= 0);
        const int src = (int)__ffsll((unsigned long long)msk) - 1;
        const int b = __shfl(bloc, src, 32);
        const int W = __shfl(Wloc, src, 32);

        // pass 2: strict winners -> global directly; boundary bin -> candidates
        for (int j = 0; j < CHUNK / 32; ++j) {
            const float v = rowS[j * 32 + lane];
            const int bin = (int)(ordu(v) >> 24);
            if (bin > b) {
                const int w = atomicAdd(&sWCnt[rowL], 1);   // w < 50 guaranteed
                outP[w] = v;
            } else if (bin == b) {
                const int ci = atomicAdd(&sCCnt[rowL], 1);
                if (ci < CANDCAP) cand[ci] = v;
            }
        }
        asm volatile("s_wait_dscnt 0" ::: "memory");
        __builtin_amdgcn_wave_barrier();

        int c = sCCnt[rowL]; if (c > CANDCAP) c = CANDCAP;
        const int need = NNEG - W;                          // >= 1, <= H[b]
        for (int it = 0; it < need; ++it) {
            float best = NEG_INF;
            int   bidx = lane;
            for (int ci = lane; ci < c; ci += 32) {
                const float v = cand[ci];
                if (v > best) { best = v; bidx = ci; }
            }
#pragma unroll
            for (int off = 16; off > 0; off >>= 1) {
                const float ov = __shfl_xor(best, off, 32);
                const int   oi = __shfl_xor(bidx, off, 32);
                if (ov > best || (ov == best && oi < bidx)) { best = ov; bidx = oi; }
            }
            if ((bidx & 31) == lane && bidx < c) cand[bidx] = NEG_INF;
            asm volatile("s_wait_dscnt 0" ::: "memory");
            __builtin_amdgcn_wave_barrier();
            if (lane == 0) outP[W + it] = best;
        }
    }
}

// =====================================================================
// K3: per-row finalize — fp32-exact positive score, merge 64*50 partial
//     candidates -> top-50, stable LSE, smoothed-CE row loss.
// =====================================================================
__global__ void __launch_bounds__(32) k_finalize(const float* __restrict__ f,
                                                 const float* __restrict__ centers,
                                                 const int* __restrict__ label,
                                                 const float* __restrict__ partial,
                                                 float* __restrict__ rowloss) {
    __shared__ float cand[NCHUNK * NNEG];   // 3200 floats
    const int row  = blockIdx.x;
    const int lane = threadIdx.x;

    const float* p = partial + (size_t)row * NCHUNK * NNEG;
    for (int i = lane; i < NCHUNK * NNEG; i += 32) cand[i] = p[i];

    const int lab = label[row];
    float acc = 0.f;
    for (int d = lane; d < D_; d += 32)
        acc += f[(size_t)row * D_ + d] * centers[(size_t)lab * D_ + d];
#pragma unroll
    for (int off = 16; off > 0; off >>= 1) acc += __shfl_xor(acc, off, 32);
    const float pos = acc;

    asm volatile("s_wait_dscnt 0" ::: "memory");
    __builtin_amdgcn_wave_barrier();

    float m = 0.f, sexp = 0.f, sneg = 0.f;
    for (int it = 0; it < NNEG; ++it) {
        float best = NEG_INF;
        int   bidx = lane;
        for (int j = 0; j < (NCHUNK * NNEG) / 32; ++j) {
            const int idx = j * 32 + lane;
            const float v = cand[idx];
            if (v > best) { best = v; bidx = idx; }
        }
#pragma unroll
        for (int off = 16; off > 0; off >>= 1) {
            const float ov = __shfl_xor(best, off, 32);
            const int   oi = __shfl_xor(bidx, off, 32);
            if (ov > best || (ov == best && oi < bidx)) { best = ov; bidx = oi; }
        }
        if ((bidx & 31) == lane) cand[bidx] = NEG_INF;
        asm volatile("s_wait_dscnt 0" ::: "memory");
        __builtin_amdgcn_wave_barrier();
        if (it == 0) { m = fmaxf(pos, best); sexp = __expf(pos - m); }
        sexp += __expf(best - m);
        sneg += best;
    }

    if (lane == 0) {
        const float lse = m + __logf(sexp);
        const float ts  = EPS_ / KSMOOTH;
        rowloss[row] = (1.0f - EPS_) * (lse - pos)
                     + ts * ((lse - pos) + ((float)NNEG * lse - sneg));
    }
}

// =====================================================================
// K4: deterministic reduction of row losses -> mean (scalar)
// =====================================================================
__global__ void __launch_bounds__(256) k_reduce(const float* __restrict__ rowloss,
                                                float* __restrict__ out) {
    __shared__ float s[256];
    const int tid = threadIdx.x;
    float a = 0.f;
    for (int i = tid; i < B_; i += 256) a += rowloss[i];
    s[tid] = a;
    __syncthreads();
    for (int w = 128; w > 0; w >>= 1) {
        if (tid < w) s[tid] += s[tid + w];
        __syncthreads();
    }
    if (tid == 0) out[0] = s[0] / (float)B_;
}

// =====================================================================
// launcher — workspace: 1MB f(bf16) | 32MB centers(bf16) | 26.2MB partial
//            | 8KB row losses  (needs ws_size >= ~59.5 MB)
// =====================================================================
extern "C" void kernel_launch(void* const* d_in, const int* in_sizes, int n_in,
                              void* d_out, int out_size, void* d_ws, size_t ws_size,
                              hipStream_t stream) {
    const float* f       = (const float*)d_in[0];
    const float* centers = (const float*)d_in[1];
    const int*   label   = (const int*)d_in[2];
    float*       out     = (float*)d_out;

    char* ws = (char*)d_ws;
    const size_t FB_BYTES   = (size_t)B_ * D_ * 2;
    const size_t CB_BYTES   = (size_t)C_ * D_ * 2;
    const size_t PART_BYTES = (size_t)B_ * NCHUNK * NNEG * 4;

    unsigned short* fb      = (unsigned short*)ws;
    unsigned short* cb      = (unsigned short*)(ws + FB_BYTES);
    float*          partial = (float*)(ws + FB_BYTES + CB_BYTES);
    float*          rowloss = (float*)(ws + FB_BYTES + CB_BYTES + PART_BYTES);

    k_convert<<<4096, 256, 0, stream>>>(f, centers, fb, cb);
    k_gemm_topk<<<dim3(NRT, NCHUNK), 256, (size_t)SMEM_K2, stream>>>(fb, cb, label, partial);
    k_finalize<<<B_, 32, 0, stream>>>(f, centers, label, partial, rowloss);
    k_reduce<<<1, 256, 0, stream>>>(rowloss, out);
}
